// AdaptiveGaussianTrendV2_3401614098930
// MI455X (gfx1250) — matrix operand order repaired and checked by reading, not compiled, not run on hardware
//
#include <hip/hip_runtime.h>
#include <hip/hip_bf16.h>

typedef __attribute__((ext_vector_type(16))) _Float16 v16h;
typedef __attribute__((ext_vector_type(8)))  _Float16 v8h;
typedef __attribute__((ext_vector_type(8)))  float    v8f;

namespace agt {
constexpr int B = 32, T = 2048, C = 64;
constexpr int TS   = 256;                 // time slab per workgroup
constexpr int HALO = 512;                 // max gaussian radius
constexpr int XT   = TS + 2 * HALO;       // 1280 staged time samples
constexpr int XSTR = 1304;                // padded LDS row stride (halves): 16B-aligned, conflict-twisted
constexpr int NS   = 5;
constexpr int   RR[NS]    = {32, 64, 128, 256, 512};
constexpr float SIG[NS]   = {8.f, 16.f, 32.f, 64.f, 128.f};
constexpr int   LL[NS]    = {65, 129, 257, 513, 1025};
constexpr int   CHK[NS]   = {3, 5, 9, 17, 33};        // ceil((L+15)/32) K-chunks
constexpr int   ABASE[NS] = {0, 3, 8, 17, 34};
constexpr int   TOFF[NS + 1] = {0, 65, 194, 451, 964, 1989};
constexpr int ATOT  = 67;
constexpr int NWAVE = 4;
constexpr int BLOCK = 32 * NWAVE;
// async staging pipeline
constexpr int ELEMS = 16 * XT;            // 20480 f32 slab elements
constexpr int CH_E  = 2048;               // elements per async chunk
constexpr int NCH   = ELEMS / CH_E;       // 10 chunks
constexpr int ISS   = CH_E / 4 / BLOCK;   // 4 b128 async issues per thread per chunk
// LDS layout (bytes)
constexpr int A_HALVES = ATOT * 32 * 16;      // banded-kernel WMMA A tiles (f16)
constexpr int X_HALVES = 16 * XSTR;           // staged x slab (f16)
constexpr int TAPS_F   = 2048;                // raw taps (f32) + inv-sums at [2000..2004]
constexpr int STAGE_F  = 2 * CH_E;            // double-buffered async f32 staging (16 KB)
constexpr int FEAT_H = 256 * 4;               // per-wave: feats [256 elems][4] f16
constexpr int HB_H   = 16 * 32;               // per-wave: hidden  [16 elems][32] f16
constexpr int WB_H   = 256 * 8;               // per-wave: logits/weights [256][8] f16
constexpr size_t OFF_A     = 0;
constexpr size_t OFF_X     = OFF_A + (size_t)A_HALVES * 2;
constexpr size_t OFF_TAP   = OFF_X + (size_t)X_HALVES * 2;
constexpr size_t OFF_STAGE = OFF_TAP + (size_t)TAPS_F * 4;
constexpr size_t OFF_WAVE  = OFF_STAGE + (size_t)STAGE_F * 4;
constexpr size_t WAVE_BYTES = (size_t)FEAT_H * 2 + (size_t)HB_H * 2 + (size_t)WB_H * 2;
constexpr size_t SMEM_TOTAL = OFF_WAVE + (size_t)NWAVE * WAVE_BYTES;   // ~160 KB
}

__device__ __forceinline__ int refl2048(int t) {
  t = (t < 0) ? -t : t;
  t = (t >= agt::T) ? (2 * agt::T - 2 - t) : t;
  return t;
}
__device__ __forceinline__ v16h cat16(v8h lo, v8h hi) {
  v16h r;
#pragma unroll
  for (int i = 0; i < 8; ++i) { r[i] = lo[i]; r[i + 8] = hi[i]; }
  return r;
}
__device__ __forceinline__ float gelu_erf(float x) {
  return 0.5f * x * (1.0f + erff(x * 0.70710678118654752f));
}
__device__ __forceinline__ v8f wmma16(v16h a, v16h b, v8f c) {
  // D = A(16x32 f16) x B(32x16 f16) + C(16x16 f32)
  return __builtin_amdgcn_wmma_f32_16x16x32_f16(false, a, false, b, (short)0, c, false, false);
}

__global__ void __launch_bounds__(agt::BLOCK)
agt_fused_kernel(const float* __restrict__ x,
                 const float* __restrict__ W1, const float* __restrict__ b1,
                 const float* __restrict__ W2, const float* __restrict__ b2,
                 const float* __restrict__ W3, const float* __restrict__ b3,
                 float* __restrict__ out)
{
  using namespace agt;
  extern __shared__ char smem[];
  _Float16* Ah   = (_Float16*)(smem + OFF_A);
  _Float16* Xh   = (_Float16*)(smem + OFF_X);
  float*    taps = (float*)   (smem + OFF_TAP);
  float*    stg  = (float*)   (smem + OFF_STAGE);

  const int tid  = threadIdx.x;
  const int wave = tid >> 5;
  const int lane = tid & 31;
  const int hlf  = lane >> 4;     // wave32 half (WMMA lane grouping)
  const int lm   = lane & 15;

  char* wsm = smem + OFF_WAVE + (size_t)wave * WAVE_BYTES;
  _Float16* feats = (_Float16*)wsm;
  _Float16* hb    = (_Float16*)(wsm + (size_t)FEAT_H * 2);
  _Float16* wb    = (_Float16*)(wsm + (size_t)FEAT_H * 2 + (size_t)HB_H * 2);

  const int slab = blockIdx.x;            // 0..T/TS-1
  const int c0   = blockIdx.y * 16;       // channel tile
  const int bb   = blockIdx.z;            // batch
  const int t0g  = slab * TS;
  const float* xb = x + (size_t)bb * T * C;

  // ---- async staged copy global->LDS (f32, b128, ASYNCcnt pipelined) ----
  // chunk i, issue slot s = k*BLOCK+tid covers 4 consecutive channels of one
  // (reflected) time sample; each wave later converts exactly the slots it
  // issued, so per-wave in-order ASYNCcnt completion is the only sync needed.
  auto issue_chunk = [&](int i) {
    const unsigned dbase = (unsigned)(uintptr_t)(stg + (i & 1) * CH_E);
#pragma unroll
    for (int k = 0; k < ISS; ++k) {
      int slot = k * BLOCK + tid;
      int idx4 = i * (CH_E / 4) + slot;
      int tl = idx4 >> 2, c4 = (idx4 & 3) * 4;
      int tg = refl2048(t0g - HALO + tl);
      const float* gp = xb + (size_t)tg * C + c0 + c4;
      unsigned daddr = dbase + (unsigned)(slot * 16);
      asm volatile("global_load_async_to_lds_b128 %0, %1, off"
                   :: "v"(daddr), "v"(gp) : "memory");
    }
  };
  auto convert_chunk = [&](int i) {
    const float* src = stg + (i & 1) * CH_E;
#pragma unroll
    for (int k = 0; k < ISS; ++k) {
      int slot = k * BLOCK + tid;
      int idx4 = i * (CH_E / 4) + slot;
      int tl = idx4 >> 2, c4 = (idx4 & 3) * 4;
      const float* p = src + slot * 4;
#pragma unroll
      for (int j = 0; j < 4; ++j)
        Xh[(c4 + j) * XSTR + tl] = (_Float16)p[j];
    }
  };

  issue_chunk(0);
  issue_chunk(1);

  // ---- raw gaussian taps (overlaps the in-flight async DMA) ----
  for (int idx = tid; idx < TOFF[NS]; idx += BLOCK) {
    int s = 0;
    while (idx >= TOFF[s + 1]) ++s;
    float n = (float)(idx - TOFF[s] - RR[s]);
    float q = n / SIG[s];
    taps[idx] = __expf(-0.5f * q * q);
  }

  for (int i = 0; i < NCH; ++i) {
    if (i + 1 < NCH) asm volatile("s_wait_asynccnt 4" ::: "memory");
    else             asm volatile("s_wait_asynccnt 0" ::: "memory");
    convert_chunk(i);
    if (i + 2 < NCH) issue_chunk(i + 2);
  }
  __syncthreads();

  if (tid < NS) {
    float ssum = 0.f;
    for (int j = 0; j < LL[tid]; ++j) ssum += taps[TOFF[tid] + j];
    taps[2000 + tid] = 1.0f / (ssum + 1e-12f);
  }
  __syncthreads();
  // ---- build Toeplitz-band A tiles in exact WMMA A-register layout ----
  // A_q(ti, K) = k[q*32 + K - ti] ; lane holds row ti=lane&15, elems map K=(e<8?e:e+8)+hlf*8
  for (int idx = tid; idx < ATOT * 32; idx += BLOCK) {
    int cs = idx >> 5, l = idx & 31;
    int s = 0;
    while (s + 1 < NS && cs >= ABASE[s + 1]) ++s;
    int q  = cs - ABASE[s];
    int ti = l & 15, hh = l >> 4;
    float inv = taps[2000 + s];
    _Float16* dst = Ah + (size_t)(cs * 32 + l) * 16;
#pragma unroll
    for (int e = 0; e < 16; ++e) {
      int K = ((e < 8) ? e : (e + 8)) + hh * 8;
      int j = q * 32 + K - ti;
      dst[e] = (_Float16)((j >= 0 && j < LL[s]) ? taps[TOFF[s] + j] * inv : 0.f);
    }
  }
  __syncthreads();

  // ---- per-lane constant MLP operands in WMMA B layout: B(K,N), N=lm(+16), K=(e<8?e:e+8)+hlf*8 ----
  v16h B1r[2], B2r[2], B3r;
#pragma unroll
  for (int t = 0; t < 2; ++t) {
    int N = lm + 16 * t;
    v16h r1, r2;
#pragma unroll
    for (int e = 0; e < 16; ++e) {
      int K = ((e < 8) ? e : (e + 8)) + hlf * 8;
      r1[e] = (K < 3) ? (_Float16)W1[N * 3 + K] : (_Float16)0.f;   // W1:[32,3]
      r2[e] = (_Float16)W2[N * 32 + K];                            // W2:[32,32]
    }
    B1r[t] = r1; B2r[t] = r2;
  }
  {
#pragma unroll
    for (int e = 0; e < 16; ++e) {
      int K = ((e < 8) ? e : (e + 8)) + hlf * 8;
      B3r[e] = (lm < 5) ? (_Float16)W3[lm * 32 + K] : (_Float16)0.f; // W3:[5,32]
    }
  }
  const float b1a = b1[lm], b1b = b1[lm + 16];
  const float b2a = b2[lm], b2b = b2[lm + 16];
  const float b3a = (lm < 5) ? b3[lm] : 0.f;

  const int TPW = (TS / 16) / NWAVE;   // t0-tiles per wave
  for (int tt = 0; tt < TPW; ++tt) {
    const int t0l = (wave * TPW + tt) * 16;

    // ---- 5-scale depthwise gaussian: banded matmul on the matrix pipe ----
    v8f acc[NS];
#pragma unroll
    for (int s = 0; s < NS; ++s) {
      v8f a = {};
      const int tbase = t0l + HALO - RR[s] + hlf * 8;    // B operand: K=(e<8?e:e+8)+hlf*8 time samples
      for (int q = 0; q < CHK[s]; ++q) {
        const v8h* ap = (const v8h*)(Ah + (size_t)((ABASE[s] + q) * 32 + lane) * 16);
        v16h av = cat16(ap[0], ap[1]);
        const v8h* xp = (const v8h*)(Xh + lm * XSTR + tbase + q * 32);
        v16h bv = cat16(xp[0], xp[2]);
        a = wmma16(av, bv, a);
      }
      acc[s] = a;
    }

    // ---- windowed local stats (win=16, reflect handled by halo) -> feats ----
#pragma unroll
    for (int v = 0; v < 8; ++v) {
      int ti = v + hlf * 8;
      int base = lm * XSTR + t0l + HALO + ti - 7;
      float s1 = 0.f, s2 = 0.f, sc = 0.f, xc = 0.f;
#pragma unroll
      for (int j = 0; j < 16; ++j) {
        float xv = (float)Xh[base + j];
        s1 += xv; s2 += xv * xv; sc += xv * ((float)j - 7.5f);
        if (j == 7) xc = xv;
      }
      float mean = s1 * (1.f / 16.f);
      float ex2  = s2 * (1.f / 16.f);
      float var  = fmaxf(ex2 - mean * mean, 0.f);
      float stdv = sqrtf(var + 1e-6f);
      float z    = fminf(fmaxf((xc - mean) / stdv, -10.f), 10.f);
      float lv   = logf(var + 1e-6f) * 0.1f;
      float slope = sc / (340.f + 1e-6f);               // t_var for win=16
      float nsl  = fminf(fmaxf(slope / (stdv + 1e-6f), -10.f), 10.f);
      _Float16* fp = feats + (size_t)(((lm << 4) | ti) << 2);
      fp[0] = (_Float16)z; fp[1] = (_Float16)lv; fp[2] = (_Float16)nsl; fp[3] = (_Float16)0.f;
    }
    asm volatile("" ::: "memory");

    // ---- conditioning MLP (WMMA) + gating logits, one channel-column group at a time ----
    for (int g = 0; g < 16; ++g) {
      v16h a1;
#pragma unroll
      for (int e = 0; e < 16; ++e) {
        int K = ((e < 8) ? e : (e + 8)) + hlf * 8;
        a1[e] = (K < 3) ? feats[(((g << 4) | lm) << 2) + K] : (_Float16)0.f;
      }
      v8f z8 = {};
      v8f h0 = wmma16(a1, B1r[0], z8);
      v8f h1 = wmma16(a1, B1r[1], z8);
#pragma unroll
      for (int v = 0; v < 8; ++v) {
        int m = v + hlf * 8;
        hb[m * 32 + lm]      = (_Float16)gelu_erf(h0[v] + b1a);
        hb[m * 32 + lm + 16] = (_Float16)gelu_erf(h1[v] + b1b);
      }
      asm volatile("" ::: "memory");
      v16h a2 = cat16(*(const v8h*)(hb + lm * 32 + hlf * 8),
                      *(const v8h*)(hb + lm * 32 + 16 + hlf * 8));
      v8f g0 = wmma16(a2, B2r[0], z8);
      v8f g1 = wmma16(a2, B2r[1], z8);
#pragma unroll
      for (int v = 0; v < 8; ++v) {
        int m = v + hlf * 8;
        hb[m * 32 + lm]      = (_Float16)gelu_erf(g0[v] + b2a);
        hb[m * 32 + lm + 16] = (_Float16)gelu_erf(g1[v] + b2b);
      }
      asm volatile("" ::: "memory");
      v16h a3 = cat16(*(const v8h*)(hb + lm * 32 + hlf * 8),
                      *(const v8h*)(hb + lm * 32 + 16 + hlf * 8));
      v8f lg = wmma16(a3, B3r, z8);
      if (lm < 5) {
#pragma unroll
        for (int v = 0; v < 8; ++v)
          wb[(((g << 4) | (v + hlf * 8)) << 3) + lm] =
              (_Float16)((lg[v] + b3a) * (1.0f / 0.7f));
      }
    }
    asm volatile("" ::: "memory");

    // ---- softmax over K=5 (f32 math, f16 storage) ----
#pragma unroll
    for (int i = 0; i < 8; ++i) {
      _Float16* p = wb + (size_t)((i * 32 + lane) << 3);
      float l0 = (float)p[0], l1 = (float)p[1], l2 = (float)p[2];
      float l3 = (float)p[3], l4 = (float)p[4];
      float m = fmaxf(fmaxf(fmaxf(l0, l1), fmaxf(l2, l3)), l4);
      float e0 = __expf(l0 - m), e1 = __expf(l1 - m), e2 = __expf(l2 - m);
      float e3 = __expf(l3 - m), e4 = __expf(l4 - m);
      float inv = 1.0f / (e0 + e1 + e2 + e3 + e4);
      p[0] = (_Float16)(e0 * inv); p[1] = (_Float16)(e1 * inv);
      p[2] = (_Float16)(e2 * inv); p[3] = (_Float16)(e3 * inv);
      p[4] = (_Float16)(e4 * inv);
    }
    asm volatile("" ::: "memory");

    // ---- gated combine (conv accumulators still live in VGPRs) + coalesced store ----
#pragma unroll
    for (int v = 0; v < 8; ++v) {
      int ti = v + hlf * 8;
      const _Float16* wp = wb + (size_t)(((lm << 4) | ti) << 3);
      float o = (float)wp[0] * acc[0][v] + (float)wp[1] * acc[1][v]
              + (float)wp[2] * acc[2][v] + (float)wp[3] * acc[3][v]
              + (float)wp[4] * acc[4][v];
      out[((size_t)bb * T + (t0g + t0l + ti)) * C + c0 + lm] = o;
    }
  }
}

extern "C" void kernel_launch(void* const* d_in, const int* in_sizes, int n_in,
                              void* d_out, int out_size, void* d_ws, size_t ws_size,
                              hipStream_t stream) {
  (void)in_sizes; (void)n_in; (void)out_size; (void)d_ws; (void)ws_size;
  const float* x  = (const float*)d_in[0];
  const float* W1 = (const float*)d_in[1];
  const float* b1 = (const float*)d_in[2];
  const float* W2 = (const float*)d_in[3];
  const float* b2 = (const float*)d_in[4];
  const float* W3 = (const float*)d_in[5];
  const float* b3 = (const float*)d_in[6];
  float* out = (float*)d_out;

  hipFuncSetAttribute(reinterpret_cast<const void*>(agt_fused_kernel),
                      hipFuncAttributeMaxDynamicSharedMemorySize,
                      (int)agt::SMEM_TOTAL);

  dim3 grid(agt::T / agt::TS, agt::C / 16, agt::B);   // 8 x 4 x 32 = 1024 WGs
  agt_fused_kernel<<<grid, dim3(agt::BLOCK), agt::SMEM_TOTAL, stream>>>(
      x, W1, b1, W2, b2, W3, b3, out);
}
